// MultiScaleRetention_40638980555017
// MI455X (gfx1250) — compile-verified, matrix operand order checked by reading, not run on hardware
//
#include <hip/hip_runtime.h>
#include <hip/hip_bf16.h>
#include <stdint.h>

// ---------------------------------------------------------------------------
// MultiScaleRetention forward for MI455X (gfx1250, wave32, WMMA bf16 path)
// B=4 T=2048 D=1024 H=8  KD=128 HD=256  SCALING=128^-0.5 (folded into Wk/bk)
// ---------------------------------------------------------------------------

typedef __attribute__((ext_vector_type(16))) __bf16 v16bf;
typedef __attribute__((ext_vector_type(8)))  __bf16 v8bf;
typedef __attribute__((ext_vector_type(8)))  float  v8f;
typedef __attribute__((ext_vector_type(4)))  float  v4f;
typedef __attribute__((ext_vector_type(4)))  int    v4i_;

#define SCALING 0.08838834764831845f   // 128^-0.5
#define EPS_GN  1e-6f

// --- CDNA5 async global->LDS copy (ASYNCcnt path), guarded --------------------
#if defined(__has_builtin)
#if __has_builtin(__builtin_amdgcn_global_load_async_to_lds_b128)
#define HAVE_ASYNC_LDS 1
#endif
#if __has_builtin(__builtin_amdgcn_s_wait_asynccnt)
#define HAVE_WAIT_ASYNC_BUILTIN 1
#endif
#endif

#ifdef HAVE_ASYNC_LDS
// Builtin signature (from clang diagnostic): params are 16-B int-vector
// pointers; global src in AS1, LDS dst in AS3, then imm offset + imm cpol.
#define ASYNC_CP16(gp, lp)                                                     \
    __builtin_amdgcn_global_load_async_to_lds_b128(                            \
        (__attribute__((address_space(1))) v4i_*)(gp),                         \
        (__attribute__((address_space(3))) v4i_*)(lp), 0, 0)
static __device__ inline void wait_async() {
#ifdef HAVE_WAIT_ASYNC_BUILTIN
    __builtin_amdgcn_s_wait_asynccnt(0);
#else
    asm volatile("s_wait_asynccnt 0" ::: "memory");
#endif
}
#endif

static __device__ inline v8f wmma_bf16(v16bf a, v16bf b, v8f c) {
    return __builtin_amdgcn_wmma_f32_16x16x32_bf16(
        /*neg_a=*/false, a, /*neg_b=*/false, b,
        /*c_mod=*/(short)0, c, /*reuse_a=*/false, /*reuse_b=*/false);
}

// Load one 16x32 bf16 A/B fragment for a lane (ISA 7.12.2 layout):
// K = half*8 + [0..7]  and  K = 16 + half*8 + [0..7].
static __device__ inline v16bf load_frag(const __bf16* p, int half) {
    v8bf lo = *(const v8bf*)(p + half * 8);
    v8bf hi = *(const v8bf*)(p + 16 + half * 8);
    v16bf r;
#pragma unroll
    for (int i = 0; i < 8; ++i) { r[i] = lo[i]; r[8 + i] = hi[i]; }
    return r;
}

// ---------------------------------------------------------------------------
// Kernel 1: weight packing / bf16 conversion.
// ---------------------------------------------------------------------------
__global__ __launch_bounds__(256) void msr_convert_weights(
    const float* __restrict__ Wq, const float* __restrict__ bq,
    const float* __restrict__ Wk, const float* __restrict__ bk,
    const float* __restrict__ Wv, const float* __restrict__ bv,
    const float* __restrict__ Wg, const float* __restrict__ bg,
    const float* __restrict__ Wo,
    __bf16* __restrict__ Wcat, float* __restrict__ bcat, __bf16* __restrict__ WoB)
{
    int64_t tid = (int64_t)blockIdx.x * blockDim.x + threadIdx.x;
    int64_t stride = (int64_t)gridDim.x * blockDim.x;
    for (int64_t i = tid; i < (int64_t)1024 * 6144; i += stride) {
        int r = (int)(i / 6144), c = (int)(i % 6144);
        float v;
        if      (c < 1024) v = Wq[(int64_t)r * 1024 + c];
        else if (c < 2048) v = Wk[(int64_t)r * 1024 + (c - 1024)] * SCALING;
        else if (c < 4096) v = Wv[(int64_t)r * 2048 + (c - 2048)];
        else               v = Wg[(int64_t)r * 2048 + (c - 4096)];
        Wcat[i] = (__bf16)v;
    }
    for (int64_t i = tid; i < (int64_t)2048 * 1024; i += stride)
        WoB[i] = (__bf16)Wo[i];
    for (int64_t i = tid; i < 6144; i += stride) {
        float v;
        if      (i < 1024) v = bq[i];
        else if (i < 2048) v = bk[i - 1024] * SCALING;
        else if (i < 4096) v = bv[i - 2048];
        else               v = bg[i - 4096];
        bcat[i] = v;
    }
}

// ---------------------------------------------------------------------------
// Kernel 2: fused QKVG projection GEMM.  C[8192,6144] = x(bf16) @ Wcat + bcat
// Block 256 thr (8 waves), 128x128x32; wave tile 64x32 (4x2 WMMA tiles).
// ---------------------------------------------------------------------------
__global__ __launch_bounds__(256) void msr_gemm_qkvg(
    const float* __restrict__ x, const __bf16* __restrict__ Wcat,
    const float* __restrict__ bcat,
    __bf16* __restrict__ qb, __bf16* __restrict__ kb,
    __bf16* __restrict__ vb, float* __restrict__ gf)
{
    __shared__ __bf16 ldsA[128 * 32];   // [m][k] row-major
    __shared__ __bf16 ldsB[128 * 32];   // [n][k] (column-major B tile)

    const int tid = threadIdx.x;
    const int wid = tid >> 5, lane = tid & 31;
    const int half = lane >> 4, lanelo = lane & 15;
    const int m0 = blockIdx.y * 128, n0 = blockIdx.x * 128;
    const int waveM = wid >> 2, waveN = wid & 3;

    v8f acc[4][2];
#pragma unroll
    for (int mt = 0; mt < 4; ++mt)
#pragma unroll
        for (int nt = 0; nt < 2; ++nt)
            acc[mt][nt] = (v8f){0, 0, 0, 0, 0, 0, 0, 0};

    for (int k0 = 0; k0 < 1024; k0 += 32) {
        if (k0 + 32 < 1024)
            __builtin_prefetch(x + (int64_t)m0 * 1024 + k0 + 32, 0, 1);
        // A: 128x32 f32 -> bf16, vectorized float4 reads (4/thread)
#pragma unroll
        for (int it = 0; it < 4; ++it) {
            int idx4 = (it * 256 + tid) * 4;
            int r = idx4 >> 5, c = idx4 & 31;
            v4f f = *(const v4f*)(x + (int64_t)(m0 + r) * 1024 + k0 + c);
#pragma unroll
            for (int i = 0; i < 4; ++i) ldsA[idx4 + i] = (__bf16)f[i];
        }
        // B: 32x128 bf16 -> [n][k] transpose, 16B reads (2/thread)
#pragma unroll
        for (int it = 0; it < 2; ++it) {
            int idx8 = (it * 256 + tid) * 8;
            int kk = idx8 >> 7, nn = idx8 & 127;
            v8bf w = *(const v8bf*)(Wcat + (int64_t)(k0 + kk) * 6144 + n0 + nn);
#pragma unroll
            for (int i = 0; i < 8; ++i) ldsB[(nn + i) * 32 + kk] = w[i];
        }
        __syncthreads();

        v16bf aF[4], bF[2];
#pragma unroll
        for (int mt = 0; mt < 4; ++mt)
            aF[mt] = load_frag(&ldsA[(waveM * 64 + mt * 16 + lanelo) * 32], half);
#pragma unroll
        for (int nt = 0; nt < 2; ++nt)
            bF[nt] = load_frag(&ldsB[(waveN * 32 + nt * 16 + lanelo) * 32], half);
#pragma unroll
        for (int mt = 0; mt < 4; ++mt)
#pragma unroll
            for (int nt = 0; nt < 2; ++nt)
                acc[mt][nt] = wmma_bf16(aF[mt], bF[nt], acc[mt][nt]);
        __syncthreads();
    }

#pragma unroll
    for (int mt = 0; mt < 4; ++mt)
#pragma unroll
        for (int nt = 0; nt < 2; ++nt)
#pragma unroll
            for (int r = 0; r < 8; ++r) {
                int m = m0 + waveM * 64 + mt * 16 + r + half * 8;
                int n = n0 + waveN * 32 + nt * 16 + lanelo;
                float val = acc[mt][nt][r] + bcat[n];
                int bb = m >> 11, t = m & 2047;      // T = 2048
                if (n < 1024) {
                    int h = n >> 7, kd = n & 127;
                    qb[((int64_t)(bb * 8 + h) * 2048 + t) * 128 + kd] = (__bf16)val;
                } else if (n < 2048) {
                    int nn = n - 1024; int h = nn >> 7, kd = nn & 127;
                    kb[((int64_t)(bb * 8 + h) * 2048 + t) * 128 + kd] = (__bf16)val;
                } else if (n < 4096) {
                    int nn = n - 2048; int h = nn >> 8, hd = nn & 255;
                    vb[((int64_t)(bb * 8 + h) * 2048 + t) * 256 + hd] = (__bf16)val;
                } else {
                    gf[(int64_t)m * 2048 + (n - 4096)] = val;
                }
            }
}

// ---------------------------------------------------------------------------
// Kernel 3: fused retention (flash-style).  Decay factored as
//   gamma^(i-j) = gamma^(i-qrow) * gamma^(qrow-j)
// so per-element work is compare + 2 mults (exp2f hoisted).
// K tile staged via global_load_async_to_lds_b128 (ASYNCcnt path).
// ---------------------------------------------------------------------------
__global__ __launch_bounds__(256) void msr_retention(
    const __bf16* __restrict__ qb, const __bf16* __restrict__ kb,
    const __bf16* __restrict__ vb, const float* __restrict__ gf,
    __bf16* __restrict__ og)
{
    __shared__ __bf16 ldsK[32 * 128];     // [j-j0][kd] row-major (contiguous copy)
    __shared__ __bf16 ldsV[256 * 32];     // [c][j-j0]  (column-major V tile)
    __shared__ __bf16 ldsP[8][16 * 32];   // per-wave C->A transpose scratch

    const int tid = threadIdx.x;
    const int wid = tid >> 5, lane = tid & 31;
    const int half = lane >> 4, lanelo = lane & 15;
    const int bh = blockIdx.y, b = bh >> 3, h = bh & 7;
    const int i0 = blockIdx.x * 128;

    const float gamma = 1.0f - exp2f(-(float)(5 + h));
    const float l2g   = log2f(gamma);

    const __bf16* qh = qb + (int64_t)bh * 2048 * 128;
    const __bf16* kh = kb + (int64_t)bh * 2048 * 128;
    const __bf16* vh = vb + (int64_t)bh * 2048 * 256;

    const int qrow = i0 + wid * 16;       // this wave's 16 query rows

    v16bf aQ[4];
#pragma unroll
    for (int kc = 0; kc < 4; ++kc)
        aQ[kc] = load_frag(qh + (int64_t)(qrow + lanelo) * 128 + kc * 32, half);

    float rowfac[8];                      // gamma^(r + half*8), r = 0..7
#pragma unroll
    for (int r = 0; r < 8; ++r)
        rowfac[r] = exp2f((float)(r + half * 8) * l2g);

    v8f accO[16];
#pragma unroll
    for (int nt = 0; nt < 16; ++nt) accO[nt] = (v8f){0, 0, 0, 0, 0, 0, 0, 0};
    float rs[8] = {0, 0, 0, 0, 0, 0, 0, 0};

    const int jmax = i0 + 127;
    for (int j0 = 0; j0 <= jmax; j0 += 32) {
        // --- K tile: 32x128 bf16 = 8KB contiguous in global -----------------
#ifdef HAVE_ASYNC_LDS
#pragma unroll
        for (int it = 0; it < 2; ++it) {              // 512 x b128, 2/thread
            int ee = it * 256 + tid;
            ASYNC_CP16(kh + (int64_t)j0 * 128 + ee * 8, &ldsK[ee * 8]);
        }
#else
#pragma unroll
        for (int it = 0; it < 2; ++it) {
            int ee = it * 256 + tid;
            *(v8bf*)&ldsK[ee * 8] = *(const v8bf*)(kh + (int64_t)j0 * 128 + ee * 8);
        }
#endif
        // --- V tile: 32x256 -> [c][j] transpose, 16B global reads -----------
#pragma unroll
        for (int it = 0; it < 4; ++it) {              // 1024 chunks, 4/thread
            int idx8 = (it * 256 + tid) * 8;
            int jj = idx8 >> 8, c0 = idx8 & 255;
            v8bf t = *(const v8bf*)(vh + (int64_t)(j0 + jj) * 256 + c0);
#pragma unroll
            for (int i = 0; i < 8; ++i) ldsV[(c0 + i) * 32 + jj] = t[i];
        }
#ifdef HAVE_ASYNC_LDS
        wait_async();
#endif
        __syncthreads();

        if (j0 <= qrow + 15) {                        // causal wave-level skip
#pragma unroll
            for (int nt = 0; nt < 2; ++nt) {
                v8f s = (v8f){0, 0, 0, 0, 0, 0, 0, 0};
#pragma unroll
                for (int kc = 0; kc < 4; ++kc)
                    s = wmma_bf16(aQ[kc],
                                  load_frag(&ldsK[(nt * 16 + lanelo) * 128 + kc * 32], half),
                                  s);
                int j = j0 + nt * 16 + lanelo;
                float colfac = exp2f((float)(qrow - j) * l2g);   // gamma^(qrow-j)
#pragma unroll
                for (int r = 0; r < 8; ++r) {
                    int d = qrow + r + half * 8 - j;
                    float w = (d >= 0) ? rowfac[r] * colfac : 0.0f;
                    float sv = s[r] * w;
                    rs[r] += sv;
                    ldsP[wid][(r + half * 8) * 32 + nt * 16 + lanelo] = (__bf16)sv;
                }
            }
            // wave-private LDS transpose: DS ops in-order per wave; wait anyway
            asm volatile("s_wait_dscnt 0" ::: "memory");
            v16bf pf = load_frag(&ldsP[wid][lanelo * 32], half);
#pragma unroll
            for (int nt = 0; nt < 16; ++nt)
                accO[nt] = wmma_bf16(pf,
                                     load_frag(&ldsV[(nt * 16 + lanelo) * 32], half),
                                     accO[nt]);
        }
        __syncthreads();
    }

    // Epilogue: row-norm + denom clamp + group-norm + SiLU gating
#pragma unroll
    for (int r = 0; r < 8; ++r) {
        int i = qrow + r + half * 8;
        float v = rs[r];
        for (int m = 1; m < 16; m <<= 1) v += __shfl_xor(v, m, 16);
        float gp = exp2f((float)(i + 1) * l2g);              // gamma^(i+1)
        float sumdecay = (1.0f - gp) / (1.0f - gamma);       // sum_{d<=i} gamma^d
        float rown = rsqrtf(sumdecay);                       // mask row normalizer
        float denom = fmaxf(fabsf(v * rown), 1.0f);
        float scale = rown / denom;

        float s1 = 0.0f, s2 = 0.0f;
#pragma unroll
        for (int nt = 0; nt < 16; ++nt) {
            float ov = accO[nt][r] * scale;
            s1 += ov; s2 += ov * ov;
        }
        for (int m = 1; m < 16; m <<= 1) {
            s1 += __shfl_xor(s1, m, 16);
            s2 += __shfl_xor(s2, m, 16);
        }
        float mu  = s1 * (1.0f / 256.0f);
        float var = s2 * (1.0f / 256.0f) - mu * mu;
        float rstd = rsqrtf(var + EPS_GN);

        int64_t rowbase = ((int64_t)b * 2048 + i) * 2048 + h * 256;
#pragma unroll
        for (int nt = 0; nt < 16; ++nt) {
            int c = nt * 16 + lanelo;
            float ov = accO[nt][r] * scale;
            float normed = (ov - mu) * rstd;
            float gv = gf[rowbase + c];
            float sig = 1.0f / (1.0f + __expf(-gv));
            og[rowbase + c] = (__bf16)(gv * sig * normed);
        }
    }
}

// ---------------------------------------------------------------------------
// Kernel 4: output projection. out[8192,1024](f32) = og(bf16) @ WoB + bo
// A tile staged via async LDS copy (rows are 64B contiguous).
// ---------------------------------------------------------------------------
__global__ __launch_bounds__(256) void msr_gemm_out(
    const __bf16* __restrict__ og, const __bf16* __restrict__ WoB,
    const float* __restrict__ bo, float* __restrict__ out)
{
    __shared__ __bf16 ldsA[128 * 32];
    __shared__ __bf16 ldsB[128 * 32];

    const int tid = threadIdx.x;
    const int wid = tid >> 5, lane = tid & 31;
    const int half = lane >> 4, lanelo = lane & 15;
    const int m0 = blockIdx.y * 128, n0 = blockIdx.x * 128;
    const int waveM = wid >> 2, waveN = wid & 3;

    v8f acc[4][2];
#pragma unroll
    for (int mt = 0; mt < 4; ++mt)
#pragma unroll
        for (int nt = 0; nt < 2; ++nt)
            acc[mt][nt] = (v8f){0, 0, 0, 0, 0, 0, 0, 0};

    for (int k0 = 0; k0 < 2048; k0 += 32) {
        if (k0 + 32 < 2048)
            __builtin_prefetch(og + (int64_t)m0 * 2048 + k0 + 32, 0, 1);
        // A: 128 rows x 32 cols bf16; each row = 4 x 16B chunks (512 chunks)
#ifdef HAVE_ASYNC_LDS
#pragma unroll
        for (int it = 0; it < 2; ++it) {
            int e = it * 256 + tid;
            int r = e >> 2, c4 = (e & 3) * 8;
            ASYNC_CP16(og + (int64_t)(m0 + r) * 2048 + k0 + c4, &ldsA[r * 32 + c4]);
        }
#else
#pragma unroll
        for (int it = 0; it < 2; ++it) {
            int e = it * 256 + tid;
            int r = e >> 2, c4 = (e & 3) * 8;
            *(v8bf*)&ldsA[r * 32 + c4] =
                *(const v8bf*)(og + (int64_t)(m0 + r) * 2048 + k0 + c4);
        }
#endif
        // B: 32x128 -> [n][k] transpose
#pragma unroll
        for (int it = 0; it < 2; ++it) {
            int idx8 = (it * 256 + tid) * 8;
            int kk = idx8 >> 7, nn = idx8 & 127;
            v8bf w = *(const v8bf*)(WoB + (int64_t)(k0 + kk) * 1024 + n0 + nn);
#pragma unroll
            for (int i = 0; i < 8; ++i) ldsB[(nn + i) * 32 + kk] = w[i];
        }
#ifdef HAVE_ASYNC_LDS
        wait_async();
#endif
        __syncthreads();

        v16bf aF[4], bF[2];
#pragma unroll
        for (int mt = 0; mt < 4; ++mt)
            aF[mt] = load_frag(&ldsA[(waveM * 64 + mt * 16 + lanelo) * 32], half);
#pragma unroll
        for (int nt = 0; nt < 2; ++nt)
            bF[nt] = load_frag(&ldsB[(waveN * 32 + nt * 16 + lanelo) * 32], half);
#pragma unroll
        for (int mt = 0; mt < 4; ++mt)
#pragma unroll
            for (int nt = 0; nt < 2; ++nt)
                acc[mt][nt] = wmma_bf16(aF[mt], bF[nt], acc[mt][nt]);
        __syncthreads();
    }

#pragma unroll
    for (int mt = 0; mt < 4; ++mt)
#pragma unroll
        for (int nt = 0; nt < 2; ++nt)
#pragma unroll
            for (int r = 0; r < 8; ++r) {
                int m = m0 + waveM * 64 + mt * 16 + r + half * 8;
                int n = n0 + waveN * 32 + nt * 16 + lanelo;
                out[(int64_t)m * 1024 + n] = acc[mt][nt][r] + bo[n];
            }
}

// ---------------------------------------------------------------------------
// Host launcher. Input order: x, mask(unused), Wq,bq, Wk,bk, Wv,bv, Wg,bg, Wo,bo
// Workspace (~185 MB): Wcat|bcat|WoB|q|k|v|g|og, all 256B aligned.
// ---------------------------------------------------------------------------
extern "C" void kernel_launch(void* const* d_in, const int* in_sizes, int n_in,
                              void* d_out, int out_size, void* d_ws, size_t ws_size,
                              hipStream_t stream) {
    (void)in_sizes; (void)n_in; (void)out_size; (void)ws_size;

    const float* x  = (const float*)d_in[0];
    // d_in[1] = decay mask [H,T,T] — recomputed analytically on-device
    const float* Wq = (const float*)d_in[2];
    const float* bq = (const float*)d_in[3];
    const float* Wk = (const float*)d_in[4];
    const float* bk = (const float*)d_in[5];
    const float* Wv = (const float*)d_in[6];
    const float* bv = (const float*)d_in[7];
    const float* Wg = (const float*)d_in[8];
    const float* bg = (const float*)d_in[9];
    const float* Wo = (const float*)d_in[10];
    const float* bo = (const float*)d_in[11];
    float* out = (float*)d_out;

    char* ws = (char*)d_ws;
    size_t off = 0;
    auto alloc = [&](size_t bytes) -> char* {
        char* p = ws + off;
        off += (bytes + 255) & ~(size_t)255;
        return p;
    };
    __bf16* Wcat = (__bf16*)alloc((size_t)1024 * 6144 * 2);
    float*  bcat = (float*) alloc((size_t)6144 * 4);
    __bf16* WoB  = (__bf16*)alloc((size_t)2048 * 1024 * 2);
    __bf16* qb   = (__bf16*)alloc((size_t)4 * 8 * 2048 * 128 * 2);
    __bf16* kb   = (__bf16*)alloc((size_t)4 * 8 * 2048 * 128 * 2);
    __bf16* vb   = (__bf16*)alloc((size_t)4 * 8 * 2048 * 256 * 2);
    float*  gf   = (float*) alloc((size_t)4 * 2048 * 2048 * 4);
    __bf16* og   = (__bf16*)alloc((size_t)4 * 2048 * 2048 * 2);

    msr_convert_weights<<<2048, 256, 0, stream>>>(
        Wq, bq, Wk, bk, Wv, bv, Wg, bg, Wo, Wcat, bcat, WoB);

    msr_gemm_qkvg<<<dim3(48, 64), 256, 0, stream>>>(x, Wcat, bcat, qb, kb, vb, gf);

    msr_retention<<<dim3(16, 32), 256, 0, stream>>>(qb, kb, vb, gf, og);

    msr_gemm_out<<<dim3(8, 64), 256, 0, stream>>>(og, WoB, bo, out);
}
